// MoeLayer_51531017617865
// MI455X (gfx1250) — compile-verified, matrix operand order checked by reading, not run on hardware
//
#include <hip/hip_runtime.h>
#include <hip/hip_bf16.h>
#include <math.h>

// ---------------------------------------------------------------------------
// MoE (top-2 of 8 experts, SwiGLU) for MI455X / gfx1250.
//   tokens T = B*S = 8192, D = 1024, H = 2048, E = 8.
// ~206 routed GFLOP vs ~11us of HBM traffic -> WMMA-bound; bf16 WMMA with
// fp32 accumulation. TILE_M=32: each weight fragment feeds two A-frags.
// Weights are pre-swizzled once per launch into fragment-ordered bf16 in the
// workspace, so every B-fragment in the hot loop is two global_load_b128 and
// zero conversions (fallback to fp32 gather if the workspace is too small).
// ---------------------------------------------------------------------------

#define T_TOKENS 8192
#define D_DIM    1024
#define H_DIM    2048
#define E_NUM    8
#define TILE_M   32            // tokens per workgroup tile (2 WMMA M-subtiles)
#define HC       128           // hidden-chunk per fused step (8 waves x 16)
#define NCHX     (D_DIM / 8)   // 16B bf16-chunks per x row
#define NCHH     (HC / 8)      // 16B bf16-chunks per h row

typedef __bf16 bf16;
typedef __attribute__((ext_vector_type(16))) __bf16 v16bf;
typedef __attribute__((ext_vector_type(8)))  __bf16 v8bf;
typedef __attribute__((ext_vector_type(8)))  float  v8f;
typedef __attribute__((ext_vector_type(4)))  float  v4f;

// ---------------------------------------------------------------------------
// Gating: one wave per token. gate = x @ Wg (E=8), top-2, softmax over top-2,
// append (token, prob) to the chosen experts' lists via atomics.
// ---------------------------------------------------------------------------
__global__ __launch_bounds__(256)
void moe_gate_kernel(const float* __restrict__ x,
                     const float* __restrict__ Wg,
                     int*   __restrict__ cnt,       // [E]
                     int*   __restrict__ idx_list,  // [E][T]
                     float* __restrict__ wgt_list)  // [E][T]
{
    const int tok  = blockIdx.x * 8 + (threadIdx.x >> 5);
    const int lane = threadIdx.x & 31;
    if (tok >= T_TOKENS) return;

    float acc[E_NUM];
#pragma unroll
    for (int e = 0; e < E_NUM; ++e) acc[e] = 0.f;

    const float* xp = x + (size_t)tok * D_DIM;
    for (int d = lane; d < D_DIM; d += 32) {
        const float xv = xp[d];
        const float* wr = Wg + (size_t)d * E_NUM;
#pragma unroll
        for (int e = 0; e < E_NUM; ++e) acc[e] += xv * wr[e];
    }
#pragma unroll
    for (int off = 16; off > 0; off >>= 1) {
#pragma unroll
        for (int e = 0; e < E_NUM; ++e)
            acc[e] += __shfl_xor(acc[e], off, 32);
    }

    if (lane == 0) {
        int i1 = 0; float v1 = acc[0];
#pragma unroll
        for (int e = 1; e < E_NUM; ++e)
            if (acc[e] > v1) { v1 = acc[e]; i1 = e; }
        int i2 = -1; float v2 = -INFINITY;
#pragma unroll
        for (int e = 0; e < E_NUM; ++e)
            if (e != i1 && acc[e] > v2) { v2 = acc[e]; i2 = e; }
        const float z  = __expf(v2 - v1);
        const float p1 = __builtin_amdgcn_rcpf(1.f + z);   // v_rcp_f32
        const float p2 = z * p1;

        int p = atomicAdd(&cnt[i1], 1);
        idx_list[i1 * T_TOKENS + p] = tok;
        wgt_list[i1 * T_TOKENS + p] = p1;
        p = atomicAdd(&cnt[i2], 1);
        idx_list[i2 * T_TOKENS + p] = tok;
        wgt_list[i2 * T_TOKENS + p] = p2;
    }
}

// ---------------------------------------------------------------------------
// Weight pre-swizzle: row-major fp32 [E][K][N] -> fragment-ordered bf16.
// Per (kblk: 32 K, nblk: 16 N) WMMA block, the 512 bf16 are stored lane-major:
// lane L owns n = nblk*16 + (L&15), kb = (L>>4)*8, elements K {kb..kb+7,
// kb+16..kb+23} as 32 contiguous bytes. A B-frag load then is 2x b128/lane
// and the wave reads 1KB contiguous.
// task = ((e*KB + kblk)*NB + nblk)*32 + lane
// ---------------------------------------------------------------------------
__global__ __launch_bounds__(256)
void convert_w_kernel(const float* __restrict__ src, bf16* __restrict__ dst,
                      int KB, int NB)
{
    const size_t task = (size_t)blockIdx.x * 256 + threadIdx.x;
    const int    lane = (int)(task & 31);
    size_t blk  = task >> 5;
    const int nblk = (int)(blk % NB); blk /= NB;
    const int kblk = (int)(blk % KB);
    const int e    = (int)(blk / KB);
    const int N = NB * 16, K = KB * 32;

    const float* s = src + ((size_t)e * K + kblk * 32 + ((lane >> 4) << 3)) * N
                         + nblk * 16 + (lane & 15);
    v8bf lo, hi;
#pragma unroll
    for (int i = 0; i < 8; ++i) {
        lo[i] = (bf16)s[(size_t)i * N];
        hi[i] = (bf16)s[(size_t)(16 + i) * N];
    }
    v8bf* d = (v8bf*)dst + task * 2;
    d[0] = lo;
    d[1] = hi;
}

// ---------------------------------------------------------------------------
// Fragment helpers (V_WMMA_F32_16X16X32_BF16 layouts, ISA 7.12.2).
// LDS tiles are 16B chunks with XOR swizzle (chunk ^ (row&15)) so the 16 rows
// of a fragment hit distinct bank groups -> conflict-free ds_load_b128.
// ---------------------------------------------------------------------------
__device__ inline v16bf load_a_swz(const v8bf* __restrict__ rowbase, int r, int kb)
{
    const int cc = kb >> 3;
    const v8bf lo = rowbase[(cc    ) ^ (r & 15)];
    const v8bf hi = rowbase[(cc + 2) ^ (r & 15)];
    return __builtin_shufflevector(lo, hi, 0,1,2,3,4,5,6,7,8,9,10,11,12,13,14,15);
}

// Fast path: fragment-ordered bf16 weights (2x global_load_b128, no cvt).
__device__ inline v16bf load_b_frag_bf16(const bf16* __restrict__ Wfe,
                                         int kblk, int nblk, int NB, int lane)
{
    const v8bf* p = (const v8bf*)Wfe + (((size_t)kblk * NB + nblk) * 32 + lane) * 2;
    return __builtin_shufflevector(p[0], p[1], 0,1,2,3,4,5,6,7,8,9,10,11,12,13,14,15);
}

// Fallback: gather from row-major fp32 [K x ld] weights, fp32->bf16 on the fly.
__device__ inline v16bf load_b_frag_f32(const float* __restrict__ W,
                                        int k0, int n, int ld)
{
    const float* p = W + (size_t)k0 * ld + n;
    v16bf r;
#pragma unroll
    for (int i = 0; i < 8; ++i) r[i]     = (bf16)p[(size_t)i * ld];
#pragma unroll
    for (int i = 0; i < 8; ++i) r[8 + i] = (bf16)p[(size_t)(16 + i) * ld];
    return r;
}

__device__ inline v8f wmma_bf16(v16bf a, v16bf b, v8f c)
{
    return __builtin_amdgcn_wmma_f32_16x16x32_bf16(
        false, a, false, b, (short)0, c, false, false);
}

// ---------------------------------------------------------------------------
// Fused expert kernel. Block = (expert e, 32-token tile of e's routed list).
// Per hidden-chunk of 128: each wave WMMAs a 32x16 slice of a = x@W1 and
// b = x@W2 (K=1024, B-frags reused across both M-subtiles), applies
// swish(a)*b -> bf16 LDS, then contracts the chunk against W3 into persistent
// 32x1024 fp32 accumulators. Rows are scaled by routing prob and atomically
// combined into out.
// ---------------------------------------------------------------------------
template <bool BF16W>
__global__ __launch_bounds__(256)
void moe_expert_kernel(const float* __restrict__ x,
                       const float* __restrict__ W1, const float* __restrict__ W2,
                       const float* __restrict__ W3,
                       const bf16*  __restrict__ W1f, const bf16* __restrict__ W2f,
                       const bf16*  __restrict__ W3f,
                       const int*   __restrict__ cnt,
                       const int*   __restrict__ idx_list,
                       const float* __restrict__ wgt_list,
                       float* __restrict__ out)
{
    const int e   = blockIdx.x;
    const int n_e = cnt[e];
    if ((int)blockIdx.y * TILE_M >= n_e) return;   // uniform early-exit

    __shared__ v8bf  xsv[TILE_M * NCHX];           // 64 KB x tile (bf16, swz)
    __shared__ v8bf  hsv[TILE_M * NCHH];           // 8 KB  h chunk (bf16, swz)
    __shared__ int   ttok[TILE_M];
    __shared__ float twgt[TILE_M];

    const int tid  = threadIdx.x;
    const int lane = tid & 31;
    const int wave = tid >> 5;                     // 0..7

    if (tid < TILE_M) {
        const int i = blockIdx.y * TILE_M + tid;
        if (i < n_e) {
            ttok[tid] = idx_list[e * T_TOKENS + i];
            twgt[tid] = wgt_list[e * T_TOKENS + i];
        } else {                                   // pad rows: weight 0
            ttok[tid] = 0;
            twgt[tid] = 0.f;
        }
    }
    __syncthreads();

    // gather + fp32->bf16 convert the 32x1024 token tile (2x b128 + cvt + b128)
    for (int idx = tid; idx < TILE_M * NCHX; idx += 256) {
        const int r  = idx >> 7;                   // / NCHX
        const int cc = idx & (NCHX - 1);
        const float* gp = x + (size_t)ttok[r] * D_DIM + (cc << 3);
        const v4f f0 = *(const v4f*)gp;
        const v4f f1 = *(const v4f*)(gp + 4);
        v8bf c;
#pragma unroll
        for (int j = 0; j < 4; ++j) { c[j] = (bf16)f0[j]; c[4 + j] = (bf16)f1[j]; }
        xsv[r * NCHX + (cc ^ (r & 15))] = c;
    }
    __syncthreads();

    const float* W1e = W1 + (size_t)e * D_DIM * H_DIM;
    const float* W2e = W2 + (size_t)e * D_DIM * H_DIM;
    const float* W3e = W3 + (size_t)e * H_DIM * D_DIM;
    const bf16*  W1fe = W1f + (size_t)e * D_DIM * H_DIM;
    const bf16*  W2fe = W2f + (size_t)e * D_DIM * H_DIM;
    const bf16*  W3fe = W3f + (size_t)e * H_DIM * D_DIM;

    const int m_row = lane & 15;                   // fragment row (per subtile)
    const int kb_a  = (lane >> 4) << 3;            // 0 or 8 (lane-half K base)
    bf16* hsf = (bf16*)hsv;

    v8f oacc0[8] = {};                             // rows 0..15  x 128 cols
    v8f oacc1[8] = {};                             // rows 16..31 x 128 cols

    for (int hc = 0; hc < H_DIM; hc += HC) {
        if (hc + HC < H_DIM) {                     // warm next chunk in L2
            if (BF16W) {
                __builtin_prefetch(W1fe + ((size_t)(hc + HC) >> 4) * 512 + tid * 64, 0, 1);
                __builtin_prefetch(W3fe + (size_t)((hc + HC) >> 5) * 64 * 512 + tid * 128, 0, 1);
            } else {
                __builtin_prefetch(W1e + (size_t)(tid >> 5) * H_DIM + hc + HC + (lane << 2), 0, 1);
                __builtin_prefetch(W3e + (size_t)(hc + HC + (tid >> 3)) * D_DIM + ((tid & 7) << 7), 0, 1);
            }
        }

        // ---- stage 1: this wave's 16 hidden columns, both M-subtiles ----
        const int bn   = hc + wave * 16 + (lane & 15);   // fallback column
        const int nblk = (hc >> 4) + wave;               // fast-path n block
        v8f a0 = {}, b0 = {}, a1 = {}, b1a = {};
        for (int k = 0; k < D_DIM; k += 32) {
            const v16bf af0 = load_a_swz(&xsv[(m_row     ) * NCHX], m_row, k + kb_a);
            const v16bf af1 = load_a_swz(&xsv[(m_row + 16) * NCHX], m_row, k + kb_a);
            v16bf w1fr, w2fr;
            if (BF16W) {
                w1fr = load_b_frag_bf16(W1fe, k >> 5, nblk, H_DIM / 16, lane);
                w2fr = load_b_frag_bf16(W2fe, k >> 5, nblk, H_DIM / 16, lane);
            } else {
                w1fr = load_b_frag_f32(W1e, k + kb_a, bn, H_DIM);
                w2fr = load_b_frag_f32(W2e, k + kb_a, bn, H_DIM);
            }
            a0  = wmma_bf16(af0, w1fr, a0);        // B-frags reused for both
            a1  = wmma_bf16(af1, w1fr, a1);        // M-subtiles
            b0  = wmma_bf16(af0, w2fr, b0);
            b1a = wmma_bf16(af1, w2fr, b1a);
        }

        __syncthreads();   // previous chunk's hs fully consumed
        {
            const int nl = wave * 16 + (lane & 15);
            const int mb = (lane >> 4) << 3;
#pragma unroll
            for (int v = 0; v < 8; ++v) {
                const int m0 = v + mb;
                const int m1 = m0 + 16;
                float a = a0[v];
                float sw = a * __builtin_amdgcn_rcpf(1.f + __expf(-a));
                hsf[(m0 * NCHH + ((nl >> 3) ^ (m0 & 15))) * 8 + (nl & 7)] =
                    (bf16)(sw * b0[v]);
                a = a1[v];
                sw = a * __builtin_amdgcn_rcpf(1.f + __expf(-a));
                hsf[(m1 * NCHH + ((nl >> 3) ^ (m1 & 15))) * 8 + (nl & 7)] =
                    (bf16)(sw * b1a[v]);
            }
        }
        __syncthreads();

        // ---- stage 2: oacc += hs[32xHC] @ W3[hc:hc+HC, wave*128:+128] ----
#pragma unroll
        for (int kk = 0; kk < HC; kk += 32) {
            const v16bf hf0 = load_a_swz(&hsv[(m_row     ) * NCHH], m_row, kk + kb_a);
            const v16bf hf1 = load_a_swz(&hsv[(m_row + 16) * NCHH], m_row, kk + kb_a);
#pragma unroll
            for (int nt = 0; nt < 8; ++nt) {
                v16bf wf;
                if (BF16W) {
                    wf = load_b_frag_bf16(W3fe, (hc + kk) >> 5, wave * 8 + nt,
                                          D_DIM / 16, lane);
                } else {
                    const int n3 = wave * 128 + nt * 16 + (lane & 15);
                    wf = load_b_frag_f32(W3e, hc + kk + kb_a, n3, D_DIM);
                }
                oacc0[nt] = wmma_bf16(hf0, wf, oacc0[nt]);  // W3 frag reused
                oacc1[nt] = wmma_bf16(hf1, wf, oacc1[nt]);  // for both subtiles
            }
        }
    }

    // ---- combine: scale rows by routing prob, atomic-add into out ----
    const int mb = (lane >> 4) << 3;
#pragma unroll
    for (int nt = 0; nt < 8; ++nt) {
        const int n = wave * 128 + nt * 16 + (lane & 15);
#pragma unroll
        for (int v = 0; v < 8; ++v) {
            const int m0 = v + mb;
            unsafeAtomicAdd(&out[(size_t)ttok[m0] * D_DIM + n],
                            oacc0[nt][v] * twgt[m0]);
            unsafeAtomicAdd(&out[(size_t)ttok[m0 + 16] * D_DIM + n],
                            oacc1[nt][v] * twgt[m0 + 16]);
        }
    }
}

// ---------------------------------------------------------------------------
// Launch. Workspace: [E cnt | E*T idx | E*T wgt | W1f | W2f | W3f(bf16)]
// Fast path needs ~101 MB of workspace; otherwise fp32-gather fallback.
// ---------------------------------------------------------------------------
extern "C" void kernel_launch(void* const* d_in, const int* in_sizes, int n_in,
                              void* d_out, int out_size, void* d_ws, size_t ws_size,
                              hipStream_t stream)
{
    const float* x  = (const float*)d_in[0];
    const float* Wg = (const float*)d_in[1];
    const float* W1 = (const float*)d_in[2];
    const float* W2 = (const float*)d_in[3];
    const float* W3 = (const float*)d_in[4];
    float* out = (float*)d_out;

    char*  ws       = (char*)d_ws;
    int*   cnt      = (int*)ws;                                   // E
    int*   idx_list = (int*)(ws + 256);                           // E*T
    float* wgt_list = (float*)(ws + 256 + (size_t)E_NUM * T_TOKENS * 4);

    const size_t roff   = 256 + 2 * (size_t)E_NUM * T_TOKENS * 4; // routing end
    const size_t wbytes = (size_t)E_NUM * D_DIM * H_DIM * 2;      // per tensor
    bf16* W1f = (bf16*)(ws + roff);
    bf16* W2f = (bf16*)(ws + roff + wbytes);
    bf16* W3f = (bf16*)(ws + roff + 2 * wbytes);
    const bool fast = ws_size >= roff + 3 * wbytes;

    hipMemsetAsync(cnt, 0, E_NUM * sizeof(int), stream);
    hipMemsetAsync(out, 0, (size_t)out_size * sizeof(float), stream);

    moe_gate_kernel<<<T_TOKENS / 8, 256, 0, stream>>>(x, Wg, cnt, idx_list, wgt_list);

    dim3 grid(E_NUM, T_TOKENS / TILE_M);   // worst case: all tokens on one expert
    if (fast) {
        // fragment-order bf16 conversion (once per launch, ~13us of HBM)
        const int g1 = E_NUM * (D_DIM / 32) * (H_DIM / 16) * 32 / 256;
        const int g3 = E_NUM * (H_DIM / 32) * (D_DIM / 16) * 32 / 256;
        convert_w_kernel<<<g1, 256, 0, stream>>>(W1, W1f, D_DIM / 32, H_DIM / 16);
        convert_w_kernel<<<g1, 256, 0, stream>>>(W2, W2f, D_DIM / 32, H_DIM / 16);
        convert_w_kernel<<<g3, 256, 0, stream>>>(W3, W3f, H_DIM / 32, D_DIM / 16);
        moe_expert_kernel<true><<<grid, 256, 0, stream>>>(
            x, W1, W2, W3, W1f, W2f, W3f, cnt, idx_list, wgt_list, out);
    } else {
        moe_expert_kernel<false><<<grid, 256, 0, stream>>>(
            x, W1, W2, W3, nullptr, nullptr, nullptr, cnt, idx_list, wgt_list, out);
    }
}
